// SelfAttention_62646392979839
// MI455X (gfx1250) — matrix operand order repaired and checked by reading, not compiled, not run on hardware
//
#include <hip/hip_runtime.h>
#include <hip/hip_bf16.h>
#include <math.h>

// Problem constants (from reference): B=2, T=2048, C=1024, H=16, D=64
#define BB 2
#define TT 2048
#define CC 1024
#define HH 16
#define DD 64

typedef __bf16 bf16;
typedef __attribute__((ext_vector_type(16))) __bf16 v16bf;
typedef __attribute__((ext_vector_type(8)))  float  v8f;
typedef __attribute__((ext_vector_type(4)))  unsigned u32x4;

union FragBF { v16bf v; u32x4 u[2]; };

static __device__ __forceinline__ unsigned short bf16_bits(float f) {
  union { float f; unsigned u; } a; a.f = f;
  unsigned r = a.u + 0x7FFFu + ((a.u >> 16) & 1u);  // round-to-nearest-even
  return (unsigned short)(r >> 16);
}
static __device__ __forceinline__ bf16 to_bf16(float f) {
  union { unsigned short s; bf16 b; } o; o.s = bf16_bits(f);
  return o.b;
}
static __device__ __forceinline__ unsigned pack_bf16x2(float x, float y) {
  return (unsigned)bf16_bits(x) | ((unsigned)bf16_bits(y) << 16);
}

static __device__ __forceinline__ float rmax16(float x) {
#pragma unroll
  for (int o = 1; o < 16; o <<= 1) x = fmaxf(x, __shfl_xor(x, o, 32));
  return x;
}
static __device__ __forceinline__ float rsum16(float x) {
#pragma unroll
  for (int o = 1; o < 16; o <<= 1) x += __shfl_xor(x, o, 32);
  return x;
}

static __device__ __forceinline__ v8f wmma_bf16(const FragBF& a, const FragBF& b, v8f c) {
  return __builtin_amdgcn_wmma_f32_16x16x32_bf16(false, a.v, false, b.v,
                                                 (short)0, c, false, false);
}

// ---------------------------------------------------------------------------
// Kernel 1: convert W (K x N, f32, row-major) -> Wt (N x K, bf16, row-major)
// so WMMA B-fragments are contiguous 16B loads per lane.
// ---------------------------------------------------------------------------
__global__ void attn_wcvt_kernel(const float* __restrict__ W, bf16* __restrict__ Wt,
                                 int K, int N) {
  int idx = blockIdx.x * blockDim.x + threadIdx.x;
  if (idx >= K * N) return;
  int k = idx / N, n = idx % N;
  Wt[(size_t)n * K + k] = to_bf16(W[idx]);
}

// ---------------------------------------------------------------------------
// Shared GEMM mainloop.
// Block = 256 threads = 8 waves arranged 2(M) x 4(N).
// Block tile 32(M) x 256(N); wave tile 16 x 64 (4 accumulators).
// K-step 64: 32x64 bf16 A tile staged in LDS (branch-free packed b128 stores).
// B addresses are formed inline from the kernel-arg base pointer (integer
// offsets only) so loads stay in the GLOBAL address space (global_load_b128,
// LOADcnt-only waits) — pointer arrays would demote them to flat.
// ---------------------------------------------------------------------------
#define GEMM_MAINLOOP(A_, Bt_, K_)                                              \
  const int tid  = threadIdx.x;                                                 \
  const int lane = tid & 31;                                                    \
  const int wv   = tid >> 5;                                                    \
  const int hi   = lane >> 4, lo = lane & 15;                                   \
  const int mi   = wv >> 2;   /* 0..1 : M half of block tile */                 \
  const int ni   = wv & 3;    /* 0..3 : N quarter            */                 \
  const int m0   = blockIdx.x * 32;                                             \
  const int nb   = blockIdx.y * 256 + ni * 64;                                  \
  const int ko   = hi * 8;                                                      \
  const int lr   = tid >> 3;        /* coop-load row 0..31  */                  \
  const int lc   = (tid & 7) * 8;   /* coop-load col (8 floats per thread) */   \
  const float* ap = (A_) + (size_t)(m0 + lr) * (K_) + lc;                       \
  bf16* asp = As + lr * 64 + lc;                                                \
  size_t bOff[4];                                                               \
  _Pragma("unroll")                                                             \
  for (int t = 0; t < 4; ++t)                                                   \
    bOff[t] = (size_t)(nb + t * 16 + lo) * (K_) + hi * 8;                       \
  v8f acc[4];                                                                   \
  _Pragma("unroll")                                                             \
  for (int t = 0; t < 4; ++t) acc[t] = (v8f){0.f,0.f,0.f,0.f,0.f,0.f,0.f,0.f};  \
  for (int k0 = 0; k0 < (K_); k0 += 64) {                                       \
    float2 f0 = *(const float2*)(ap + 0);                                       \
    float2 f1 = *(const float2*)(ap + 2);                                       \
    float2 f2 = *(const float2*)(ap + 4);                                       \
    float2 f3 = *(const float2*)(ap + 6);                                       \
    u32x4 pk;                                                                   \
    pk.x = pack_bf16x2(f0.x, f0.y);                                             \
    pk.y = pack_bf16x2(f1.x, f1.y);                                             \
    pk.z = pack_bf16x2(f2.x, f2.y);                                             \
    pk.w = pack_bf16x2(f3.x, f3.y);                                             \
    *(u32x4*)asp = pk;                                                          \
    ap += 64;                                                                   \
    if (k0 + 64 < (K_)) __builtin_prefetch(ap, 0, 1);                           \
    __syncthreads();                                                            \
    /* preload everything, then compute */                                      \
    FragBF af[2], bfr[2][4];                                                    \
    const bf16* arow = As + (mi * 16 + lo) * 64 + ko;                           \
    af[0].u[0] = *(const u32x4*)(arow);                                         \
    af[0].u[1] = *(const u32x4*)(arow + 16);                                    \
    af[1].u[0] = *(const u32x4*)(arow + 32);                                    \
    af[1].u[1] = *(const u32x4*)(arow + 48);                                    \
    _Pragma("unroll")                                                           \
    for (int kk2 = 0; kk2 < 2; ++kk2) {                                         \
      _Pragma("unroll")                                                         \
      for (int t = 0; t < 4; ++t) {                                             \
        const bf16* bptr = (Bt_) + (bOff[t] + (size_t)k0 + kk2 * 32);           \
        bfr[kk2][t].u[0] = *(const u32x4*)(bptr);                               \
        bfr[kk2][t].u[1] = *(const u32x4*)(bptr + 16);                          \
      }                                                                         \
    }                                                                           \
    _Pragma("unroll")                                                           \
    for (int kk2 = 0; kk2 < 2; ++kk2) {                                         \
      _Pragma("unroll")                                                         \
      for (int t = 0; t < 4; ++t)                                               \
        acc[t] = wmma_bf16(af[kk2], bfr[kk2][t], acc[t]);                       \
    }                                                                           \
    __syncthreads();                                                            \
  }

// ---------------------------------------------------------------------------
// Kernel 2a: generic WMMA GEMM, fp32 output with bias.
// Cout(MxN) = A(MxK,f32) * Bt(NxK,bf16)^T + bias
// ---------------------------------------------------------------------------
__global__ __launch_bounds__(256) void attn_gemm_bf16_kernel(
    const float* __restrict__ A, const bf16* __restrict__ Bt,
    const float* __restrict__ bias, float* __restrict__ Cout,
    int N, int K) {
  __shared__ __align__(16) bf16 As[32 * 64];  // 4 KiB
  GEMM_MAINLOOP(A, Bt, K)
#pragma unroll
  for (int t = 0; t < 4; ++t) {
    const int n = nb + t * 16 + lo;
    const float bn = bias[n];
#pragma unroll
    for (int v = 0; v < 8; ++v) {
      const int m = m0 + mi * 16 + v + 8 * hi;
      Cout[(size_t)m * N + n] = acc[t][v] + bn;
    }
  }
}

// ---------------------------------------------------------------------------
// Kernel 2b: QKV GEMM with fused scatter epilogue.
// Writes bf16 Q(B,H,T,D) scaled by 1/sqrt(D), K(B,H,T,D), V^T(B,H,D,T)
// directly from accumulators (no fp32 staging buffer).
// ---------------------------------------------------------------------------
__global__ __launch_bounds__(256) void attn_qkv_gemm_kernel(
    const float* __restrict__ A, const bf16* __restrict__ Bt,
    const float* __restrict__ bias,
    bf16* __restrict__ Qb, bf16* __restrict__ Kb, bf16* __restrict__ Vt,
    int K) {
  __shared__ __align__(16) bf16 As[32 * 64];  // 4 KiB
  GEMM_MAINLOOP(A, Bt, K)
#pragma unroll
  for (int t = 0; t < 4; ++t) {
    const int n  = nb + t * 16 + lo;       // 0..3071 (q/k/v boundaries are x1024)
    const float bn = bias[n];
    const int which = n >> 10;             // 0=q 1=k 2=v
    const int cn = n & (CC - 1);
    const int h  = cn >> 6;                // cn / D
    const int d  = cn & (DD - 1);
#pragma unroll
    for (int v = 0; v < 8; ++v) {
      const int m  = m0 + mi * 16 + v + 8 * hi;  // global row = b*T + t
      const int b_ = m >> 11;                    // / T
      const int tt = m & (TT - 1);
      const size_t bh = (size_t)(b_ * HH + h);
      const float r = acc[t][v] + bn;
      if (which == 0)      Qb[(bh * TT + tt) * DD + d] = to_bf16(r * 0.125f);
      else if (which == 1) Kb[(bh * TT + tt) * DD + d] = to_bf16(r);
      else                 Vt[(bh * DD + d) * TT + tt] = to_bf16(r);
    }
  }
}

// ---------------------------------------------------------------------------
// Kernel 3: causal flash attention. One wave per 16-row Q tile.
// Per 32-key chunk: preload 4 K-frags -> 4 S WMMAs; V-frag loads issued before
// the softmax VALU block so exp/shfl work hides their latency; P bounced
// C/D->A layout through per-wave LDS (s_wait_dscnt); 4 P.V WMMAs.
// Chunks fully below the diagonal skip the causal compare.
// ---------------------------------------------------------------------------
__global__ __launch_bounds__(128) void attn_flash_kernel(
    const bf16* __restrict__ Qb, const bf16* __restrict__ Kb,
    const bf16* __restrict__ Vt, const float* __restrict__ amask,
    float* __restrict__ Y) {
  __shared__ __align__(16) bf16 Pl[4][16][32];
  const int tid  = threadIdx.x;
  const int lane = tid & 31, wv = tid >> 5;
  const int hi   = lane >> 4, lo = lane & 15;
  const int h    = blockIdx.y, b = blockIdx.z;
  const int q0   = (blockIdx.x * 4 + wv) * 16;
  const size_t bh = (size_t)(b * HH + h);
  const bf16* qp = Qb + bh * TT * DD;
  const bf16* kp = Kb + bh * TT * DD;
  const bf16* vp = Vt + bh * DD * TT;
  const float* mp = amask + (size_t)b * TT;

  // Resident Q fragments (K-dim 64 -> two 16x32 A-frags)
  FragBF aq0, aq1;
  {
    const bf16* r = qp + (size_t)(q0 + lo) * DD + hi * 16;
    aq0.u[0] = *(const u32x4*)(r);
    aq0.u[1] = *(const u32x4*)(r + 8);
    aq1.u[0] = *(const u32x4*)(r + 32);
    aq1.u[1] = *(const u32x4*)(r + 40);
  }

  v8f o0 = {0.f,0.f,0.f,0.f,0.f,0.f,0.f,0.f};
  v8f o1 = o0, o2 = o0, o3 = o0;
  float mrow[8], lrow[8], alpha[8];
#pragma unroll
  for (int v = 0; v < 8; ++v) { mrow[v] = -INFINITY; lrow[v] = 0.f; }

  const int nch = (q0 + 16 + 31) >> 5;  // causal: keys 0 .. q0+15
  for (int ch = 0; ch < nch; ++ch) {
    const int key0 = ch * 32;
    const bool diag = (key0 + 31 > q0);  // chunk touches the diagonal?

    // ---- preload 4 K fragments, then 4 S WMMAs back-to-back ----
    FragBF kb[4];
    const bf16* kr = kp + (size_t)(key0 + lo) * DD + hi * 16;
    kb[0].u[0] = *(const u32x4*)(kr);            kb[0].u[1] = *(const u32x4*)(kr + 8);
    kb[1].u[0] = *(const u32x4*)(kr + 32);       kb[1].u[1] = *(const u32x4*)(kr + 40);
    const bf16* kr2 = kr + 16 * DD;
    kb[2].u[0] = *(const u32x4*)(kr2);           kb[2].u[1] = *(const u32x4*)(kr2 + 8);
    kb[3].u[0] = *(const u32x4*)(kr2 + 32);      kb[3].u[1] = *(const u32x4*)(kr2 + 40);
    if (ch + 1 < nch) __builtin_prefetch(kr + 32 * DD, 0, 1);  // next chunk K rows

    v8f s0 = {0.f,0.f,0.f,0.f,0.f,0.f,0.f,0.f};
    v8f s1 = s0;
    s0 = wmma_bf16(aq0, kb[0], s0);
    s0 = wmma_bf16(aq1, kb[1], s0);
    s1 = wmma_bf16(aq0, kb[2], s1);
    s1 = wmma_bf16(aq1, kb[3], s1);

    // ---- V fragments: issue loads now; softmax VALU hides their latency ----
    FragBF bv[4];
    const bf16* vr = vp + (size_t)lo * TT + key0 + hi * 8;
#pragma unroll
    for (int dt = 0; dt < 4; ++dt) {
      bv[dt].u[0] = *(const u32x4*)(vr + dt * 16 * TT);
      bv[dt].u[1] = *(const u32x4*)(vr + dt * 16 * TT + 16);
    }

    // ---- mask + online softmax ----
    const int col0 = key0 + lo, col1 = col0 + 16;
    const float pm0 = mp[col0], pm1 = mp[col1];
#pragma unroll
    for (int v = 0; v < 8; ++v) {
      float a = s0[v] + pm0;
      float c = s1[v] + pm1;
      if (diag) {
        const int row = q0 + v + hi * 8;
        if (col0 > row) a = -INFINITY;
        if (col1 > row) c = -INFINITY;
      }
      float mnew = fmaxf(mrow[v], rmax16(fmaxf(a, c)));
      float al = __expf(mrow[v] - mnew);
      float e0 = __expf(a - mnew);
      float e1 = __expf(c - mnew);
      s0[v] = e0; s1[v] = e1;
      lrow[v] = lrow[v] * al + rsum16(e0 + e1);
      mrow[v] = mnew;
      alpha[v] = al;
    }
#pragma unroll
    for (int v = 0; v < 8; ++v) {
      o0[v] *= alpha[v]; o1[v] *= alpha[v];
      o2[v] *= alpha[v]; o3[v] *= alpha[v];
    }

    // ---- P: C/D layout -> A-frag layout via per-wave LDS tile ----
#pragma unroll
    for (int v = 0; v < 8; ++v) {
      Pl[wv][v + hi * 8][lo]      = to_bf16(s0[v]);
      Pl[wv][v + hi * 8][16 + lo] = to_bf16(s1[v]);
    }
    asm volatile("s_wait_dscnt 0" ::: "memory");  // order DS stores before reads
    FragBF pf;
    pf.u[0] = *(const u32x4*)(&Pl[wv][lo][hi * 8]);
    pf.u[1] = *(const u32x4*)(&Pl[wv][lo][16 + hi * 8]);

    // ---- O += P V : 4 d-tiles of 16 (V already resident) ----
    o0 = wmma_bf16(pf, bv[0], o0);
    o1 = wmma_bf16(pf, bv[1], o1);
    o2 = wmma_bf16(pf, bv[2], o2);
    o3 = wmma_bf16(pf, bv[3], o3);
  }

  // ---- normalize and store fp32 attention output (B,T,C) ----
#pragma unroll
  for (int v = 0; v < 8; ++v) {
    float inv = 1.0f / lrow[v];
    int t = q0 + v + hi * 8;
    float* yr = Y + ((size_t)(b * TT + t)) * CC + h * DD + lo;
    yr[0]  = o0[v] * inv;
    yr[16] = o1[v] * inv;
    yr[32] = o2[v] * inv;
    yr[48] = o3[v] * inv;
  }
}

// ---------------------------------------------------------------------------
// Host-side orchestration
// ---------------------------------------------------------------------------
extern "C" void kernel_launch(void* const* d_in, const int* in_sizes, int n_in,
                              void* d_out, int out_size, void* d_ws, size_t ws_size,
                              hipStream_t stream) {
  (void)in_sizes; (void)n_in; (void)out_size; (void)ws_size;
  const float* x      = (const float*)d_in[0];   // (B,T,C) f32
  const float* amask  = (const float*)d_in[1];   // (B,T) f32 additive
  // d_in[2] = attention_bias (int tril) -> causality computed analytically
  const float* W_attn = (const float*)d_in[3];   // (C,3C) f32
  const float* b_attn = (const float*)d_in[4];   // (3C,) f32
  const float* W_proj = (const float*)d_in[5];   // (C,C) f32
  const float* b_proj = (const float*)d_in[6];   // (C,) f32
  float* out = (float*)d_out;                    // (B,T,C) f32

  char* ws = (char*)d_ws;                        // 16B-aligned offsets
  bf16*  WattnT = (bf16*)(ws);                   // 3072*1024 bf16 =  6 MiB
  bf16*  WprojT = (bf16*)(ws + 6291456);         // 1024*1024 bf16 =  2 MiB
  bf16*  Qb     = (bf16*)(ws + 8388608);         // B*H*T*D bf16   =  8 MiB
  bf16*  Kb     = (bf16*)(ws + 16777216);        //                =  8 MiB
  bf16*  Vt     = (bf16*)(ws + 25165824);        // transposed V   =  8 MiB
  float* Yat    = (float*)(ws + 33554432);       // 4096*1024 f32  = 16 MiB

  // 1) weights -> transposed bf16
  attn_wcvt_kernel<<<(CC * 3 * CC + 255) / 256, 256, 0, stream>>>(W_attn, WattnT, CC, 3 * CC);
  attn_wcvt_kernel<<<(CC * CC + 255) / 256, 256, 0, stream>>>(W_proj, WprojT, CC, CC);

  // 2) QKV GEMM (4096 x 3072 x 1024) with fused bf16 Q/K/V^T scatter epilogue
  attn_qkv_gemm_kernel<<<dim3(BB * TT / 32, 3 * CC / 256), 256, 0, stream>>>(
      x, WattnT, b_attn, Qb, Kb, Vt, CC);

  // 3) causal flash attention -> Yat (B,T,C) f32
  attn_flash_kernel<<<dim3(TT / 16 / 4, HH, BB), 128, 0, stream>>>(Qb, Kb, Vt, amask, Yat);

  // 4) output projection: out = Yat * W_proj + b_proj  (4096 x 1024 x 1024)
  attn_gemm_bf16_kernel<<<dim3(BB * TT / 32, CC / 256), 256, 0, stream>>>(
      Yat, WprojT, b_proj, out, CC, CC);
}
